// CoupledLSTM_21801253994969
// MI455X (gfx1250) — matrix operand order repaired and checked by reading, not compiled, not run on hardware
//
#include <hip/hip_runtime.h>
#include <stdint.h>
#include <stddef.h>

// Problem constants (match reference): T=512, B=32, I=H=1024
#define T_STEPS 512
#define BATCH   32
#define HID     1024
#define GATE4   (4 * HID)

typedef __bf16 bf16;
typedef __attribute__((ext_vector_type(16))) __bf16 v16bf;
typedef __attribute__((ext_vector_type(8)))  __bf16 v8bf;
typedef __attribute__((ext_vector_type(8)))  float  v8f;

// ---------------------------------------------------------------------------
// fp32 -> bf16 round-to-nearest-even (bit-level; no reliance on __bf16 conv)
// ---------------------------------------------------------------------------
static __device__ inline bf16 f2bf(float f) {
    union { float f; uint32_t u; } v;
    v.f = f;
    uint32_t r = v.u + 0x7FFFu + ((v.u >> 16) & 1u);
    unsigned short h = (unsigned short)(r >> 16);
    return __builtin_bit_cast(bf16, h);
}

// ---------------------------------------------------------------------------
// WMMA fragment loaders (contiguous k order k = 16h + s, same in A and B, wave32)
//
// A (16x32 MxK, 16-bit): lane L holds row M=L&15.
//   lanes 0-15 : elems 0..15 -> K=kBase+0..15
//   lanes 16-31: elems 0..15 -> K=kBase+16..31
//
// B (32x16 KxN, 16-bit): lane L holds col N=L&15.
//   lanes 0-15 : elems 0..15 -> K=kBase+0..15
//   lanes 16-31: elems 0..15 -> K=kBase+16..31
// Since gates = act @ Wt  (B[k][n] = W[n][k]) and W is row-major [N][K],
// each lane reads 16 *contiguous* bf16 (32B) — no transpose needed.
//
// C/D (16x16 f32): lane L holds col N=L&15; vgpr r -> row M = r + 8*(L>=16).
// ---------------------------------------------------------------------------
static __device__ inline v16bf load_a_frag(const bf16* __restrict__ A, int lda,
                                           int mBase, int kBase, int lane) {
    int m  = mBase + (lane & 15);
    int kb = (lane & 16) ? 16 : 0;
    const bf16* base = A + (size_t)m * lda + kBase;
    return *(const v16bf*)(base + kb);
}

static __device__ inline v16bf load_b_frag(const bf16* __restrict__ W, int ldw,
                                           int nBase, int kBase, int lane) {
    int n  = nBase + (lane & 15);
    int kb = (lane & 16) ? 16 : 0;
    return *(const v16bf*)(W + (size_t)n * ldw + kBase + kb);
}

static __device__ inline v8f wmma_bf16(v16bf a, v16bf b, v8f c) {
    return __builtin_amdgcn_wmma_f32_16x16x32_bf16(false, a, false, b,
                                                   (short)0, c, false, false);
}

// ---------------------------------------------------------------------------
// Elementwise fp32 -> bf16 conversion
// ---------------------------------------------------------------------------
__global__ void k_f32_to_bf16(const float* __restrict__ src,
                              bf16* __restrict__ dst, int n) {
    int i = blockIdx.x * blockDim.x + threadIdx.x;
    if (i < n) dst[i] = f2bf(src[i]);
}

// ---------------------------------------------------------------------------
// Big GEMM: C[M,N] = A[M,K](bf16) * W[N,K](bf16)^T, fp32 accumulate.
// One wave computes a 32x64 output tile (2 m-tiles x 4 n-tiles, 8 accums):
// per K-step 12 x b128 loads feed 8 WMMAs, and each weight fragment is
// shared by both m-tiles (halves L2 weight traffic vs a 16-row strip).
// Unconditional speculative prefetch of the weight stream one K-block
// ahead keeps the loop body branch-free (OOB prefetch is silently dropped).
// Grid must be exactly (M/32)*(N/64)/8 blocks of 256 threads.
// ---------------------------------------------------------------------------
__global__ void k_gemm_bf16(const bf16* __restrict__ A,
                            const bf16* __restrict__ W,
                            float* __restrict__ C, int M, int N, int K) {
    int lane = threadIdx.x & 31;
    int wid  = blockIdx.x * (blockDim.x >> 5) + (threadIdx.x >> 5);
    int wavesPerRow = N >> 6;
    int mBase = (wid / wavesPerRow) << 5;
    int nBase = (wid % wavesPerRow) << 6;

    v8f acc[2][4];
#pragma unroll
    for (int m = 0; m < 2; ++m)
#pragma unroll
        for (int j = 0; j < 4; ++j) acc[m][j] = (v8f){};

    for (int k = 0; k < K; k += 32) {
        // speculative prefetch of the next weight K-block (global_prefetch_b8)
        __builtin_prefetch(W + (size_t)(nBase + (lane & 15)) * K + k + 32, 0, 1);
        v16bf a0 = load_a_frag(A, K, mBase,      k, lane);
        v16bf a1 = load_a_frag(A, K, mBase + 16, k, lane);
        v16bf b[4];
#pragma unroll
        for (int j = 0; j < 4; ++j)
            b[j] = load_b_frag(W, K, nBase + 16 * j, k, lane);
#pragma unroll
        for (int j = 0; j < 4; ++j) {
            acc[0][j] = wmma_bf16(a0, b[j], acc[0][j]);
            acc[1][j] = wmma_bf16(a1, b[j], acc[1][j]);
        }
    }
    int n = lane & 15;
#pragma unroll
    for (int m = 0; m < 2; ++m) {
        int mOff = mBase + 16 * m + ((lane & 16) ? 8 : 0);
#pragma unroll
        for (int i = 0; i < 8; ++i) {
            float* row = C + (size_t)(mOff + i) * N + nBase + n;
            row[0]  = acc[m][0][i];
            row[16] = acc[m][1][i];
            row[32] = acc[m][2][i];
            row[48] = acc[m][3][i];
        }
    }
}

// ---------------------------------------------------------------------------
// Per-step gate pre-activations:
//   gates[B,4H] = xg_t + h @ w_hh^T  (+ c @ w_pi^T for gate i, + c @ w_pf^T
//   for gate f). 256 waves: wave = gate(2b) x ntile(6b). Launch <<<32,256>>>.
// ---------------------------------------------------------------------------
__global__ void k_gates(const float* __restrict__ xg_t,  // [B,4H]
                        const bf16*  __restrict__ hbf,   // [B,H]
                        const bf16*  __restrict__ cbf,   // [B,H]
                        const bf16*  __restrict__ whh,   // [4H,H]
                        const bf16*  __restrict__ wpi,   // [H,H]
                        const bf16*  __restrict__ wpf,   // [H,H]
                        float* __restrict__ gates)       // [B,4H]
{
    int lane  = threadIdx.x & 31;
    int wid   = blockIdx.x * (blockDim.x >> 5) + (threadIdx.x >> 5);
    int gate  = wid >> 6;
    int nLoc  = (wid & 63) << 4;
    int nGlob = gate * HID + nLoc;

    v8f acc0, acc1;
    {   // init from precomputed input projection, in C-fragment layout
        int n    = nGlob + (lane & 15);
        int mOff = (lane & 16) ? 8 : 0;
#pragma unroll
        for (int i = 0; i < 8; ++i) {
            acc0[i] = xg_t[(size_t)(mOff + i)      * GATE4 + n];
            acc1[i] = xg_t[(size_t)(mOff + i + 16) * GATE4 + n];
        }
    }
    for (int k = 0; k < HID; k += 32) {
        v16bf b  = load_b_frag(whh, HID, nGlob, k, lane);
        v16bf a0 = load_a_frag(hbf, HID, 0,  k, lane);
        v16bf a1 = load_a_frag(hbf, HID, 16, k, lane);
        acc0 = wmma_bf16(a0, b, acc0);
        acc1 = wmma_bf16(a1, b, acc1);
    }
    if (gate < 2) {  // wave-uniform branch: peephole on i / f gates
        const bf16* wp = (gate == 0) ? wpi : wpf;
        for (int k = 0; k < HID; k += 32) {
            v16bf b  = load_b_frag(wp, HID, nLoc, k, lane);
            v16bf a0 = load_a_frag(cbf, HID, 0,  k, lane);
            v16bf a1 = load_a_frag(cbf, HID, 16, k, lane);
            acc0 = wmma_bf16(a0, b, acc0);
            acc1 = wmma_bf16(a1, b, acc1);
        }
    }
    int n    = nGlob + (lane & 15);
    int mOff = (lane & 16) ? 8 : 0;
#pragma unroll
    for (int i = 0; i < 8; ++i) {
        gates[(size_t)(mOff + i)      * GATE4 + n] = acc0[i];
        gates[(size_t)(mOff + i + 16) * GATE4 + n] = acc1[i];
    }
}

// ---------------------------------------------------------------------------
// Cell update: c_new = sigmoid(gi)*tanh(gg) + sigmoid(gf)*c. <<<128,256>>>.
// ---------------------------------------------------------------------------
__global__ void k_cell(const float* __restrict__ gates,  // [B,4H]
                       float* __restrict__ c32,          // [B,H] in/out
                       bf16*  __restrict__ cbf)          // [B,H] out
{
    int i = blockIdx.x * blockDim.x + threadIdx.x;  // 0..B*H-1
    int b = i >> 10, n = i & (HID - 1);
    const float* g = gates + (size_t)b * GATE4;
    float gi = g[n], gf = g[HID + n], gg = g[2 * HID + n];
    float iv = 1.0f / (1.0f + __expf(-gi));
    float fv = 1.0f / (1.0f + __expf(-gf));
    float gv = tanhf(gg);
    float cn = fv * c32[i] + iv * gv;
    c32[i] = cn;
    cbf[i] = f2bf(cn);
}

// ---------------------------------------------------------------------------
// Output: o = gates_o + c_new @ w_po^T  (NO sigmoid, per reference);
//         h = o * tanh(c_new). 64 waves. Launch <<<8,256>>>.
// ---------------------------------------------------------------------------
__global__ void k_out(const float* __restrict__ gates,  // [B,4H] (o at 3H)
                      const bf16*  __restrict__ cbf,    // c_new bf16 [B,H]
                      const float* __restrict__ c32,    // c_new f32 [B,H]
                      const bf16*  __restrict__ wpo,    // [H,H]
                      float* __restrict__ h32,          // [B,H] state
                      bf16*  __restrict__ hbf,          // [B,H] state
                      bf16*  __restrict__ ybf,          // [B,H] slice (next-layer input)
                      float* __restrict__ y32)          // [B,H] slice or nullptr
{
    int lane = threadIdx.x & 31;
    int wid  = blockIdx.x * (blockDim.x >> 5) + (threadIdx.x >> 5);  // 0..63
    int nLoc = wid << 4;

    v8f acc0, acc1;
    {
        int n    = 3 * HID + nLoc + (lane & 15);
        int mOff = (lane & 16) ? 8 : 0;
#pragma unroll
        for (int i = 0; i < 8; ++i) {
            acc0[i] = gates[(size_t)(mOff + i)      * GATE4 + n];
            acc1[i] = gates[(size_t)(mOff + i + 16) * GATE4 + n];
        }
    }
    for (int k = 0; k < HID; k += 32) {
        v16bf b  = load_b_frag(wpo, HID, nLoc, k, lane);
        v16bf a0 = load_a_frag(cbf, HID, 0,  k, lane);
        v16bf a1 = load_a_frag(cbf, HID, 16, k, lane);
        acc0 = wmma_bf16(a0, b, acc0);
        acc1 = wmma_bf16(a1, b, acc1);
    }
    int n    = nLoc + (lane & 15);
    int mOff = (lane & 16) ? 8 : 0;
#pragma unroll
    for (int i = 0; i < 8; ++i) {
        {
            int m = mOff + i;
            float cn = c32[(size_t)m * HID + n];
            float h  = acc0[i] * tanhf(cn);
            h32[(size_t)m * HID + n] = h;
            bf16 hb = f2bf(h);
            hbf[(size_t)m * HID + n] = hb;
            ybf[(size_t)m * HID + n] = hb;
            if (y32) y32[(size_t)m * HID + n] = h;
        }
        {
            int m = mOff + i + 16;
            float cn = c32[(size_t)m * HID + n];
            float h  = acc1[i] * tanhf(cn);
            h32[(size_t)m * HID + n] = h;
            bf16 hb = f2bf(h);
            hbf[(size_t)m * HID + n] = hb;
            ybf[(size_t)m * HID + n] = hb;
            if (y32) y32[(size_t)m * HID + n] = h;
        }
    }
}

// ---------------------------------------------------------------------------
// Zero the recurrent state. <<<128,256>>>.
// ---------------------------------------------------------------------------
__global__ void k_zero_state(float* __restrict__ h32, float* __restrict__ c32,
                             bf16* __restrict__ hbf, bf16* __restrict__ cbf) {
    int i = blockIdx.x * blockDim.x + threadIdx.x;
    if (i < BATCH * HID) {
        h32[i] = 0.0f;
        c32[i] = 0.0f;
        bf16 z = __builtin_bit_cast(bf16, (unsigned short)0);
        hbf[i] = z;
        cbf[i] = z;
    }
}

// Copy final (h, c) of a layer into the hn/cn slices of d_out. <<<128,256>>>.
__global__ void k_copy_state(const float* __restrict__ h32,
                             const float* __restrict__ c32,
                             float* __restrict__ hn, float* __restrict__ cn) {
    int i = blockIdx.x * blockDim.x + threadIdx.x;
    if (i < BATCH * HID) { hn[i] = h32[i]; cn[i] = c32[i]; }
}

// ---------------------------------------------------------------------------
// Host orchestration
// ---------------------------------------------------------------------------
extern "C" void kernel_launch(void* const* d_in, const int* in_sizes, int n_in,
                              void* d_out, int out_size, void* d_ws, size_t ws_size,
                              hipStream_t stream) {
    const float* x = (const float*)d_in[0];
    const float* w32[2][5] = {
        { (const float*)d_in[1], (const float*)d_in[2], (const float*)d_in[3],
          (const float*)d_in[4], (const float*)d_in[5] },   // ih0 hh0 pi0 pf0 po0
        { (const float*)d_in[6], (const float*)d_in[7], (const float*)d_in[8],
          (const float*)d_in[9], (const float*)d_in[10] },  // ih1 hh1 pi1 pf1 po1
    };

    // ---- workspace carve-out ----
    char*  ws  = (char*)d_ws;
    size_t off = 0;
    auto carve = [&](size_t bytes) -> char* {
        char* p = ws + off;
        off += (bytes + 255) & ~(size_t)255;
        return p;
    };
    bf16* wbf[2][5];
    const size_t wElems[5] = { (size_t)GATE4 * HID, (size_t)GATE4 * HID,
                               (size_t)HID * HID, (size_t)HID * HID,
                               (size_t)HID * HID };
    for (int l = 0; l < 2; ++l)
        for (int w = 0; w < 5; ++w)
            wbf[l][w] = (bf16*)carve(sizeof(bf16) * wElems[w]);

    const size_t MROWS = (size_t)T_STEPS * BATCH;             // 16384
    bf16*  xbf   = (bf16*)carve(sizeof(bf16) * MROWS * HID);  // x in bf16
    bf16*  ybf   = (bf16*)carve(sizeof(bf16) * MROWS * HID);  // layer output bf16
    float* xg    = (float*)carve(sizeof(float) * MROWS * GATE4);
    float* gates = (float*)carve(sizeof(float) * BATCH * GATE4);
    float* h32   = (float*)carve(sizeof(float) * BATCH * HID);
    float* c32   = (float*)carve(sizeof(float) * BATCH * HID);
    bf16*  hbf   = (bf16*)carve(sizeof(bf16) * BATCH * HID);
    bf16*  cbf   = (bf16*)carve(sizeof(bf16) * BATCH * HID);

    auto cvt = [&](const float* s, bf16* d, size_t n) {
        k_f32_to_bf16<<<(unsigned)((n + 255) / 256), 256, 0, stream>>>(s, d, (int)n);
    };

    // ---- convert weights + x to bf16 ----
    for (int l = 0; l < 2; ++l)
        for (int w = 0; w < 5; ++w)
            cvt(w32[l][w], wbf[l][w], wElems[w]);
    cvt(x, xbf, MROWS * HID);

    const size_t YN = MROWS * HID;  // elements in y1
    float* out_y  = (float*)d_out;
    float* out_hn = out_y + YN;
    float* out_cn = out_hn + 2 * (size_t)BATCH * HID;

    const unsigned gemmBlocks = (unsigned)(((MROWS / 32) * (GATE4 / 64)) / 8);  // 4096
    const unsigned stateBlocks = (BATCH * HID) / 256;                           // 128

    for (int layer = 0; layer < 2; ++layer) {
        // Big input-projection GEMM for all timesteps at once.
        const bf16* Ain = (layer == 0) ? xbf : ybf;
        k_gemm_bf16<<<gemmBlocks, 256, 0, stream>>>(Ain, wbf[layer][0], xg,
                                                    (int)MROWS, GATE4, HID);
        k_zero_state<<<stateBlocks, 256, 0, stream>>>(h32, c32, hbf, cbf);

        for (int t = 0; t < T_STEPS; ++t) {
            const float* xgt = xg + (size_t)t * BATCH * GATE4;
            k_gates<<<32, 256, 0, stream>>>(xgt, hbf, cbf, wbf[layer][1],
                                            wbf[layer][2], wbf[layer][3], gates);
            k_cell<<<stateBlocks, 256, 0, stream>>>(gates, c32, cbf);
            bf16*  yslab = ybf + (size_t)t * BATCH * HID;  // layer1 input / scratch
            float* y32   = (layer == 1) ? (out_y + (size_t)t * BATCH * HID) : nullptr;
            k_out<<<8, 256, 0, stream>>>(gates, cbf, c32, wbf[layer][4],
                                         h32, hbf, yslab, y32);
        }
        k_copy_state<<<stateBlocks, 256, 0, stream>>>(
            h32, c32,
            out_hn + (size_t)layer * BATCH * HID,
            out_cn + (size_t)layer * BATCH * HID);
    }
}
